// GinNet_64991445123381
// MI455X (gfx1250) — matrix operand order repaired and checked by reading, not compiled, not run on hardware
//
#include <hip/hip_runtime.h>
#include <hip/hip_bf16.h>

#define NN 100000
#define EE 3200000
#define GG 512
#define CC 6

typedef __attribute__((ext_vector_type(16))) __bf16 v16bf;
typedef __attribute__((ext_vector_type(8)))  float  v8f;
typedef __attribute__((ext_vector_type(8)))  unsigned short us8;   // 16 bytes

union Frag { unsigned short us[16]; v16bf v; us8 h[2]; };

// native RTNE float->bf16 (lets clang emit v_cvt_*_bf16_f32 on gfx1250)
__device__ __forceinline__ unsigned short f2bf(float f) {
  __bf16 b = (__bf16)f;
  return __builtin_bit_cast(unsigned short, b);
}

__device__ __forceinline__ void pack4(unsigned short* o, float4 xv, float4 av, float e1) {
  o[0] = f2bf(e1 * xv.x + av.x);
  o[1] = f2bf(e1 * xv.y + av.y);
  o[2] = f2bf(e1 * xv.z + av.z);
  o[3] = f2bf(e1 * xv.w + av.w);
}

// ---------------- zero agg + per-layer BN stats ----------------
__global__ void k_zero(float* __restrict__ agg, float* __restrict__ stats, int n64) {
  int i = blockIdx.x * blockDim.x + threadIdx.x;
  if (i < n64) agg[i] = 0.0f;
  if (i < 128) stats[i] = 0.0f;
}

// ---------------- edge scatter-add: agg[dst] += x[src] ----------------
// 16 threads / edge, float4 gather, 4 f32 atomics into L2-resident agg.
__global__ void k_scatter(const float* __restrict__ x, const int* __restrict__ src,
                          const int* __restrict__ dst, float* __restrict__ agg) {
  int idx = blockIdx.x * blockDim.x + threadIdx.x;
  if (idx >= EE * 16) return;
  int e  = idx >> 4;
  int c4 = (idx & 15) << 2;
  int s = src[e], d = dst[e];
  const float4 v = *(const float4*)(x + s * 64 + c4);
  float* o = agg + d * 64 + c4;
  atomicAdd(o + 0, v.x); atomicAdd(o + 1, v.y);
  atomicAdd(o + 2, v.z); atomicAdd(o + 3, v.w);
}

// ---------------- fused GIN MLP: u = relu(relu(((1+eps)x+agg)@Wa+ba)@Wb+bb) ----------------
// 128 threads = 4 waves; each wave owns 16 rows -> one 16x64 output tile via
// v_wmma_f32_16x16x32_bf16 (2 K-chunks x 4 col-tiles per GEMM). Weights staged in
// LDS pre-swizzled into B-fragment order (2x ds_load_b128 per fragment). Per-column
// sum/sum^2 for BatchNorm reduced in LDS then 128 global atomics per block.
// u may alias agg (each wave reads only its own agg rows before writing them).
__global__ void __launch_bounds__(128)
k_mlp(const float* __restrict__ x, const float* __restrict__ agg,
      const float* __restrict__ wa, const float* __restrict__ ba,
      const float* __restrict__ wb, const float* __restrict__ bb,
      const float* __restrict__ epsp,
      float* __restrict__ u, float* __restrict__ stats) {
  __shared__ unsigned short wa_sw[4096];   // [kc][nt][lane][idx] fragment order
  __shared__ unsigned short wb_sw[4096];
  __shared__ unsigned short t_s[4][16 * 64];
  __shared__ float ba_s[64], bb_s[64];
  __shared__ float csum[64], csq[64];

  const int tid = threadIdx.x;
  // stage weights into exact B-fragment layout:
  //   k -> kc=k>>5, kk=k&31, hi=(kk>>3)&1, idx=(kk&7)+8*(kk>>4) ; n -> nt=n>>4, lane=16*hi+(n&15)
  for (int i = tid; i < 4096; i += 128) {
    int k = i >> 6, n = i & 63;
    int kc = k >> 5, kk = k & 31;
    int hiK = (kk >> 3) & 1, idx = (kk & 7) + ((kk >> 4) << 3);
    int nt = n >> 4;
    int ln = hiK * 16 + (n & 15);
    int pos = (((kc * 4 + nt) * 32) + ln) * 16 + idx;
    wa_sw[pos] = f2bf(wa[i]);
    wb_sw[pos] = f2bf(wb[i]);
  }
  if (tid < 64) { ba_s[tid] = ba[tid]; bb_s[tid] = bb[tid]; csum[tid] = 0.f; csq[tid] = 0.f; }
  __syncthreads();

  const float e1 = 1.0f + epsp[0];
  const int wave = tid >> 5, lane = tid & 31;
  const int hi = lane >> 4, lm = lane & 15;
  const int r0 = blockIdx.x * 64 + wave * 16;

  v8f acc[4];
  for (int nt = 0; nt < 4; ++nt) for (int j = 0; j < 8; ++j) acc[nt][j] = 0.f;

  // ---- GEMM 1: t = relu(h @ Wa + ba), h = (1+eps)x + agg, built on the fly ----
  const int rowA = r0 + lm;
  const int rowL = (rowA < NN) ? rowA : 0;         // clamp: loads unconditional
  const float* xr = x   + rowL * 64;
  const float* ar = agg + rowL * 64;
  for (int kc = 0; kc < 2; ++kc) {
    const int Kb = kc * 32;
    const int b0 = Kb + hi * 8;                    // A.us[0..7]  <- h[b0 .. b0+7]
    const int b1 = b0 + 16;                        // A.us[8..15] <- h[b1 .. b1+7]
    float4 x0 = *(const float4*)(xr + b0);
    float4 x1 = *(const float4*)(xr + b0 + 4);
    float4 x2 = *(const float4*)(xr + b1);
    float4 x3 = *(const float4*)(xr + b1 + 4);
    float4 a0 = *(const float4*)(ar + b0);
    float4 a1 = *(const float4*)(ar + b0 + 4);
    float4 a2 = *(const float4*)(ar + b1);
    float4 a3 = *(const float4*)(ar + b1 + 4);
    Frag A;
    pack4(A.us + 0,  x0, a0, e1);
    pack4(A.us + 4,  x1, a1, e1);
    pack4(A.us + 8,  x2, a2, e1);
    pack4(A.us + 12, x3, a3, e1);
    for (int nt = 0; nt < 4; ++nt) {
      Frag B;
      const us8* bp = (const us8*)&wa_sw[(((kc * 4 + nt) * 32) + lane) * 16];
      B.h[0] = bp[0]; B.h[1] = bp[1];
      acc[nt] = __builtin_amdgcn_wmma_f32_16x16x32_bf16(false, A.v, false, B.v,
                                                        (short)0, acc[nt], false, false);
    }
  }
  // bias + relu, stash t (bf16) into this wave's LDS tile
  for (int nt = 0; nt < 4; ++nt) {
    int col = nt * 16 + lm;
    float bv = ba_s[col];
    for (int j = 0; j < 8; ++j) {
      float v = acc[nt][j] + bv;
      v = v > 0.f ? v : 0.f;
      int m = j + hi * 8;                 // C/D layout: vgpr j -> row j + 8*hi, col = lm+16*nt
      t_s[wave][m * 64 + col] = f2bf(v);
    }
  }
  __syncthreads();

  // ---- GEMM 2: u = relu(t @ Wb + bb) ----
  v8f acc2[4];
  for (int nt = 0; nt < 4; ++nt) for (int j = 0; j < 8; ++j) acc2[nt][j] = 0.f;
  for (int kc = 0; kc < 2; ++kc) {
    const int Kb = kc * 32;
    Frag A;
    A.h[0] = *(const us8*)&t_s[wave][lm * 64 + Kb + hi * 8];
    A.h[1] = *(const us8*)&t_s[wave][lm * 64 + Kb + 16 + hi * 8];
    for (int nt = 0; nt < 4; ++nt) {
      Frag B;
      const us8* bp = (const us8*)&wb_sw[(((kc * 4 + nt) * 32) + lane) * 16];
      B.h[0] = bp[0]; B.h[1] = bp[1];
      acc2[nt] = __builtin_amdgcn_wmma_f32_16x16x32_bf16(false, A.v, false, B.v,
                                                         (short)0, acc2[nt], false, false);
    }
  }
  // bias + relu + store u + BN stats; wave-uniform fast path when all 16 rows valid
  if (r0 + 16 <= NN) {
    for (int nt = 0; nt < 4; ++nt) {
      int col = nt * 16 + lm;
      float bv = bb_s[col];
      for (int j = 0; j < 8; ++j) {
        float v = acc2[nt][j] + bv;
        v = v > 0.f ? v : 0.f;
        int row = r0 + j + hi * 8;
        u[row * 64 + col] = v;
        atomicAdd(&csum[col], v);
        atomicAdd(&csq[col], v * v);
      }
    }
  } else {
    for (int nt = 0; nt < 4; ++nt) {
      int col = nt * 16 + lm;
      float bv = bb_s[col];
      for (int j = 0; j < 8; ++j) {
        float v = acc2[nt][j] + bv;
        v = v > 0.f ? v : 0.f;
        int row = r0 + j + hi * 8;
        if (row < NN) {
          u[row * 64 + col] = v;
          atomicAdd(&csum[col], v);
          atomicAdd(&csq[col], v * v);
        }
      }
    }
  }
  __syncthreads();
  if (tid < 64) {
    atomicAdd(&stats[tid],      csum[tid]);
    atomicAdd(&stats[64 + tid], csq[tid]);
  }
}

// ---------------- BN finalize: fold gamma/beta into scale/shift ----------------
__global__ void k_bnfin(const float* __restrict__ stats, const float* __restrict__ gamma,
                        const float* __restrict__ beta, float* __restrict__ bnp) {
  int c = threadIdx.x;
  if (c < 64) {
    const float inv_n = 1.0f / (float)NN;
    float mu   = stats[c] * inv_n;
    float var  = stats[64 + c] * inv_n - mu * mu;   // biased var, matches jnp.var
    float istd = rsqrtf(var + 1e-5f);
    float sc   = istd * gamma[c];
    bnp[c]      = sc;
    bnp[64 + c] = beta[c] - mu * sc;
  }
}

__global__ void k_bnapply(const float* __restrict__ u, const float* __restrict__ bnp,
                          float* __restrict__ y) {
  int i = blockIdx.x * blockDim.x + threadIdx.x;
  if (i < NN * 64) {
    int c = i & 63;
    y[i] = u[i] * bnp[c] + bnp[64 + c];
  }
}

// ---------------- graph pooling ----------------
__global__ void k_poolzero(float* __restrict__ msum, int* __restrict__ mmax,
                           float* __restrict__ cnt) {
  int i = blockIdx.x * blockDim.x + threadIdx.x;
  if (i < GG * 64) { msum[i] = 0.f; mmax[i] = (int)0xFF800000; }  // -inf bits
  if (i < GG) cnt[i] = 0.f;
}

__global__ void k_poolacc(const float* __restrict__ h, const int* __restrict__ batch,
                          float* __restrict__ msum, int* __restrict__ mmax,
                          float* __restrict__ cnt) {
  int i = blockIdx.x * blockDim.x + threadIdx.x;
  if (i >= NN * 64) return;
  int node = i >> 6, col = i & 63;
  int g = batch[node];
  float v = h[i];
  atomicAdd(&msum[g * 64 + col], v);
  int* addr = &mmax[g * 64 + col];
  // order-preserving float max via int max (v>=0) / uint min (v<0)
  if (v >= 0.f) atomicMax(addr, __float_as_int(v));
  else          atomicMin((unsigned int*)addr, __float_as_uint(v));
  if (col == 0) atomicAdd(&cnt[g], 1.0f);
}

// ---------------- per-graph FC + log_softmax ----------------
__global__ void k_final(const float* __restrict__ msum, const int* __restrict__ mmax,
                        const float* __restrict__ cnt, const float* __restrict__ wfc,
                        const float* __restrict__ bfc, float* __restrict__ out) {
  __shared__ float pooled[128];
  __shared__ float logits[8];
  int g = blockIdx.x;
  int lane = threadIdx.x;
  float c = cnt[g];
  for (int col = lane; col < 64; col += 32) {
    pooled[col] = msum[g * 64 + col] / fmaxf(c, 1.0f);
    float mx = __int_as_float(mmax[g * 64 + col]);
    pooled[64 + col] = (c > 0.f) ? mx : 0.0f;
  }
  __syncthreads();
  if (lane < CC) {
    float s = bfc[lane];
    for (int k = 0; k < 128; ++k) s += pooled[k] * wfc[k * CC + lane];
    logits[lane] = s;
  }
  __syncthreads();
  if (lane < CC) {
    float mx = logits[0];
    for (int k = 1; k < CC; ++k) mx = fmaxf(mx, logits[k]);
    float se = 0.f;
    for (int k = 0; k < CC; ++k) se += expf(logits[k] - mx);
    out[g * CC + lane] = logits[lane] - mx - logf(se);
  }
}

extern "C" void kernel_launch(void* const* d_in, const int* in_sizes, int n_in,
                              void* d_out, int out_size, void* d_ws, size_t ws_size,
                              hipStream_t stream) {
  (void)in_sizes; (void)n_in; (void)out_size; (void)ws_size;
  const float* x     = (const float*)d_in[0];
  const int*   ei    = (const int*)d_in[1];
  const int*   batch = (const int*)d_in[2];
  const float* W[3][4];                           // [layer][wa,ba,wb,bb]
  for (int l = 0; l < 3; ++l)
    for (int j = 0; j < 4; ++j)
      W[l][j] = (const float*)d_in[3 + l * 4 + j];
  const float *EPS[4], *GAM[4], *BET[4];
  for (int l = 0; l < 4; ++l) {
    EPS[l] = (const float*)d_in[15 + l * 3];
    GAM[l] = (const float*)d_in[16 + l * 3];
    BET[l] = (const float*)d_in[17 + l * 3];
  }
  const float* wfc = (const float*)d_in[27];
  const float* bfc = (const float*)d_in[28];
  float* out = (float*)d_out;

  float* ws    = (float*)d_ws;
  float* agg   = ws;                     // N*64 (also holds u in place)
  float* xbuf  = agg + NN * 64;          // N*64
  float* stats = xbuf + NN * 64;         // 128
  float* bnp   = stats + 128;            // 128
  float* msum  = bnp + 128;              // G*64
  float* cnt   = msum + GG * 64;         // G
  int*   mmax  = (int*)(cnt + GG);       // G*64

  const int* src = ei;
  const int* dst = ei + EE;
  const int n64 = NN * 64;

  const float* xin = x;
  for (int l = 0; l < 4; ++l) {
    int wl = (l < 3) ? l : 2;            // conv4 reuses conv3's MLP weights
    k_zero   <<<(n64 + 255) / 256, 256, 0, stream>>>(agg, stats, n64);
    k_scatter<<<(EE * 16 + 255) / 256, 256, 0, stream>>>(xin, src, dst, agg);
    k_mlp    <<<(NN + 63) / 64, 128, 0, stream>>>(xin, agg, W[wl][0], W[wl][1],
                                                  W[wl][2], W[wl][3], EPS[l], agg, stats);
    k_bnfin  <<<1, 64, 0, stream>>>(stats, GAM[l], BET[l], bnp);
    k_bnapply<<<(n64 + 255) / 256, 256, 0, stream>>>(agg, bnp, xbuf);
    xin = xbuf;
  }

  k_poolzero<<<(GG * 64 + 255) / 256, 256, 0, stream>>>(msum, mmax, cnt);
  k_poolacc <<<(n64 + 255) / 256, 256, 0, stream>>>(xbuf, batch, msum, mmax, cnt);
  k_final   <<<GG, 32, 0, stream>>>(msum, mmax, cnt, wfc, bfc, out);
}
